// EfficientAttention_37864431681745
// MI455X (gfx1250) — matrix operand order repaired and checked by reading, not compiled, not run on hardware
//
#include <hip/hip_runtime.h>

// ---------------- problem constants ----------------
#define BB   2
#define SS   2048
#define HH   1024
#define NHH  16
#define HDD  64
#define TOPK 64
#define SSP  (SS + 4)    // padded f32 score row stride (2052 = 4 mod 64 words -> conflict-free)

typedef __attribute__((ext_vector_type(16))) _Float16 v16h;
typedef __attribute__((ext_vector_type(8)))  _Float16 v8h;
typedef __attribute__((ext_vector_type(4)))  _Float16 v4h;
typedef __attribute__((ext_vector_type(8)))  float    v8f;
typedef __attribute__((ext_vector_type(4)))  float    f32x4;
typedef __attribute__((ext_vector_type(4)))  unsigned int u32x4;
typedef __attribute__((ext_vector_type(8)))  int      i32x8;
typedef __attribute__((ext_vector_type(4)))  int      i32x4;

#if defined(__has_builtin)
#if __has_builtin(__builtin_amdgcn_tensor_load_to_lds) && __has_builtin(__builtin_amdgcn_s_wait_tensorcnt)
#define HAVE_TDM 1
#endif
#endif
#ifndef HAVE_TDM
#define HAVE_TDM 0
#endif

// ---------------- WMMA helper ----------------
__device__ __forceinline__ v8f wmma16x16x32(v16h a, v16h b, v8f c) {
  return __builtin_amdgcn_wmma_f32_16x16x32_f16(
      /*neg_a=*/false, a, /*neg_b=*/false, b,
      /*c_mod=*/(short)0, c, /*reuse_a=*/false, /*reuse_b=*/false);
}

// A-matrix 16x32 f16 fragment from a row-major tile (lda halves, rows 16B aligned).
__device__ __forceinline__ v16h load_a_frag(const _Float16* __restrict__ A, int lda) {
  int lane = threadIdx.x & 31;
  int m  = lane & 15;
  int h16 = (lane >> 4) * 16;
  const _Float16* p = A + m * lda + h16;
  v8h lo = *(const v8h*)p;          // K = h16 .. h16+7
  v8h hi = *(const v8h*)(p + 8);    // K = h16+8 .. h16+15
  return __builtin_shufflevector(lo, hi, 0,1,2,3,4,5,6,7,8,9,10,11,12,13,14,15);
}

#if HAVE_TDM
// TDM: 2D tile of f16 (tile_d0 elems/row, tile_d1 rows) global->LDS, optional LDS row padding.
// pad codes: pad_interval (1<<pi units of 8B), pad_amount ((pa+1) DWORDs); 0/0 -> no padding.
__device__ __forceinline__ void tdm_load_tile_f16(unsigned ldsOff, const _Float16* g,
                                                  unsigned tensor_d0, unsigned tensor_d1,
                                                  unsigned tile_d0, unsigned tile_d1,
                                                  unsigned d0_stride,
                                                  unsigned pi, unsigned pa, unsigned pad_en) {
  unsigned long long ga = (unsigned long long)(size_t)(const void*)g;
  u32x4 g0;
  g0[0] = 1u;                                        // count=1, user mode
  g0[1] = ldsOff;                                    // lds_addr (bytes)
  g0[2] = (unsigned)(ga & 0xFFFFFFFFu);              // global_addr[31:0]
  g0[3] = (unsigned)((ga >> 32) & 0x01FFFFFFu) | (2u << 30);  // addr[56:32] | type=2
  i32x8 g1;
  g1[0] = (int)((1u << 16) | (pad_en << 20) | (pi << 22) | (pa << 25)); // data_size=2B
  g1[1] = (int)((tensor_d0 & 0xFFFFu) << 16);                          // dim0[15:0]
  g1[2] = (int)(((tensor_d0 >> 16) & 0xFFFFu) | ((tensor_d1 & 0xFFFFu) << 16));
  g1[3] = (int)(((tensor_d1 >> 16) & 0xFFFFu) | ((tile_d0 & 0xFFFFu) << 16));
  g1[4] = (int)(tile_d1 & 0xFFFFu);                                    // tile_dim1; tile_dim2=0
  g1[5] = (int)d0_stride;                                              // dim0_stride[31:0]
  g1[6] = 0;                                                           // stride hi, dim1_stride lo
  g1[7] = 0;
  i32x4 z4 = (i32x4)0;
#if __clang_major__ >= 23
  i32x8 z8 = (i32x8)0;
  __builtin_amdgcn_tensor_load_to_lds(g0, g1, z4, z4, z8, 0);
#else
  __builtin_amdgcn_tensor_load_to_lds(g0, g1, z4, z4, 0);
#endif
}
#endif

// ---------------- wave reductions (wave32) ----------------
__device__ __forceinline__ float wred_sum(float v) {
#pragma unroll
  for (int o = 16; o > 0; o >>= 1) v += __shfl_xor(v, o, 32);
  return v;
}
__device__ __forceinline__ float wred_max(float v) {
#pragma unroll
  for (int o = 16; o > 0; o >>= 1) v = fmaxf(v, __shfl_xor(v, o, 32));
  return v;
}
__device__ __forceinline__ int wred_isum(int v) {
#pragma unroll
  for (int o = 16; o > 0; o >>= 1) v += __shfl_xor(v, o, 32);
  return v;
}

// ---------------- fp32 -> fp16 convert (vectorized, n % 4 == 0) ----------------
__global__ void cvt_f16_kernel(const float* __restrict__ src, _Float16* __restrict__ dst, int n4) {
  for (int i = blockIdx.x * blockDim.x + threadIdx.x; i < n4; i += gridDim.x * blockDim.x) {
    f32x4 x = ((const f32x4*)src)[i];
    v4h y;
#pragma unroll
    for (int q = 0; q < 4; ++q) y[q] = (_Float16)x[q];
    ((v4h*)dst)[i] = y;
  }
}

// ---------------- V transpose: V16[b*S+s][hcol] -> Vt[(b*H+hcol)*S + s] ----------------
__global__ void transpose_v_kernel(const _Float16* __restrict__ V, _Float16* __restrict__ Vt) {
  __shared__ _Float16 tile[64][72];
  const int b  = blockIdx.z;
  const int s0 = blockIdx.x * 64;
  const int c0 = blockIdx.y * 64;
  const int t  = threadIdx.x;     // 256 threads
  {
    int r = t >> 2, cc = (t & 3) * 16;
    const _Float16* g = V + ((size_t)(b * SS + s0 + r)) * HH + c0 + cc;
    *(v8h*)&tile[r][cc]     = *(const v8h*)g;
    *(v8h*)&tile[r][cc + 8] = *(const v8h*)(g + 8);
  }
  __syncthreads();
  {
    int i = t >> 2, jj = (t & 3) * 16;
    v8h o0, o1;
#pragma unroll
    for (int q = 0; q < 8; ++q) { o0[q] = tile[jj + q][i]; o1[q] = tile[jj + 8 + q][i]; }
    _Float16* o = Vt + ((size_t)b * HH + c0 + i) * SS + s0 + jj;
    *(v8h*)o       = o0;
    *(v8h*)(o + 8) = o1;
  }
}

// ---------------- tiled WMMA GEMM: out = (A[MxK] @ W[KxN] + bias) * scale ----------------
// 256 threads = 8 waves; block tile 64x128; wave tile 16x64; double-buffered k-loop.
// A slab staged by TDM (padded tile descriptor); B staged in fragment order via registers.
__global__ void gemm_f16_kernel(const _Float16* __restrict__ A,
                                const _Float16* __restrict__ W,
                                const float* __restrict__ bias,
                                float scale,
                                _Float16* __restrict__ out16,
                                float* __restrict__ out32,
                                int M, int N, int K) {
  __shared__ _Float16 As[2][64][40];         // 64 rows x 32 K-halves (+8 pad)
  __shared__ _Float16 Bsw[2][8][33][16];     // fragment-order B, padded lane dim

  const int tid  = threadIdx.x;
  const int wave = tid >> 5;
  const int lane = tid & 31;
  const int wr = wave >> 1;               // 0..3
  const int wc = wave & 1;                // 0..1
  const int row0 = blockIdx.x * 64;
  const int col0 = blockIdx.y * 128;

  // staging coordinates
  const int ar = tid >> 2, ac0 = (tid & 3) * 8;        // A: 16B per thread
  const int br = tid >> 4, bc0 = (tid & 15) * 8;       // B: k=br & br+16, 8 n's
  const int ntb = bc0 >> 4, n0 = bc0 & 15;

  v8f acc[4];
#pragma unroll
  for (int t = 0; t < 4; ++t) acc[t] = (v8f)0.0f;

  v8h wa, w0, w1;
  auto issue = [&](int kk) {
#if !HAVE_TDM
    wa = *(const v8h*)&A[(size_t)(row0 + ar) * K + kk + ac0];
#endif
    w0 = *(const v8h*)&W[(size_t)(kk + br) * N + col0 + bc0];
    w1 = *(const v8h*)&W[(size_t)(kk + br + 16) * N + col0 + bc0];
  };
  auto commit = [&](int buf) {
#if !HAVE_TDM
    *(v8h*)&As[buf][ar][ac0] = wa;
#endif
#pragma unroll
    for (int h = 0; h < 8; ++h) {
      Bsw[buf][ntb][n0 + h][br]      = w0[h];   // k = br      -> lane = n
      Bsw[buf][ntb][16 + n0 + h][br] = w1[h];   // k = br + 16 -> lane = 16 + n
    }
  };
#if HAVE_TDM
  auto tdmA = [&](int buf, int kk) {
    // 64-row x 32-half tile; LDS rows padded 64B+16B -> pad_interval=3 (8*8B), pad_amount=3 (4 DW)
    tdm_load_tile_f16((unsigned)(size_t)(void*)&As[buf][0][0],
                      &A[(size_t)row0 * K + kk],
                      (unsigned)K, (unsigned)M, 32u, 64u, (unsigned)K, 3u, 3u, 1u);
  };
#endif

  // prologue: stage k-step 0 into buffer 0
#if HAVE_TDM
  if (wave == 0) tdmA(0, 0);
#endif
  issue(0);
  commit(0);
#if HAVE_TDM
  if (wave == 0) __builtin_amdgcn_s_wait_tensorcnt(0);
#endif
  __syncthreads();

  for (int kk = 0, it = 0; kk < K; kk += 32, ++it) {
    const int cur = it & 1, nxt = cur ^ 1;
    const bool pre = (kk + 32 < K);
    if (pre) {
#if HAVE_TDM
      if (wave == 0) tdmA(nxt, kk + 32);
#endif
      issue(kk + 32);          // global loads overlap the WMMAs below
    }

    v16h a = load_a_frag(&As[cur][wr * 16][0], 40);
#pragma unroll
    for (int nt = 0; nt < 4; ++nt) {
      v16h b = *(const v16h*)&Bsw[cur][wc * 4 + nt][lane][0];   // 32B contiguous
      acc[nt] = wmma16x16x32(a, b, acc[nt]);
    }

    if (pre) commit(nxt);
#if HAVE_TDM
    if (wave == 0) __builtin_amdgcn_s_wait_tensorcnt(0);
#endif
    __syncthreads();
  }

  // epilogue: (acc + bias) * scale
  const int half = lane >> 4;
  const int nloc = lane & 15;
#pragma unroll
  for (int nt = 0; nt < 4; ++nt) {
#pragma unroll
    for (int r = 0; r < 8; ++r) {
      int row = row0 + wr * 16 + r + 8 * half;
      int col = col0 + wc * 64 + nt * 16 + nloc;
      float v = (acc[nt][r] + bias[col]) * scale;
      if (out16) out16[(size_t)row * N + col] = (_Float16)v;
      if (out32) out32[(size_t)row * N + col] = v;
    }
  }
}

// ---------------- attention with exact top-k threshold ----------------
// grid: BB*NHH*(SS/16) blocks; 512 threads = 16 waves.
// dynamic LDS: sc[16*SSP] f32 | Qs[16*72] f16 | Ol[16*64] f32 | rsum[16] f32
#define SC_BYTES   (16 * SSP * 4)
#define QS_BYTES   (16 * 72 * 2)
#define OL_BYTES   (16 * 64 * 4)
#define ATTN_SMEM_BYTES (SC_BYTES + QS_BYTES + OL_BYTES + 16 * 4)

__global__ void attn_topk_kernel(const _Float16* __restrict__ Q,
                                 const _Float16* __restrict__ Kt,
                                 const _Float16* __restrict__ Vt,
                                 _Float16* __restrict__ O16) {
  extern __shared__ char smem[];
  float*    sc   = (float*)smem;                                    // 16 x SSP
  _Float16* Qs   = (_Float16*)(smem + SC_BYTES);                    // 16 x 72
  float*    Ol   = (float*)(smem + SC_BYTES + QS_BYTES);            // 16 x 64
  float*    rsum = (float*)(smem + SC_BYTES + QS_BYTES + OL_BYTES); // 16

  const int tile = blockIdx.x;
  const int qt   = tile % (SS / 16);
  const int head = (tile / (SS / 16)) % NHH;
  const int b    = tile / ((SS / 16) * NHH);
  const int s0   = qt * 16;

  const int tid  = threadIdx.x;
  const int wave = tid >> 5;         // 0..15
  const int lane = tid & 31;
  const int half = lane >> 4;
  const int nloc = lane & 15;
  const int h16  = half * 16;
  const int h8   = half * 8;

  // ---- phase 0: stage Q tile, zero output accumulator ----
#if HAVE_TDM
  if (wave == 0) {
    // 16-row x 64-half tile; rows padded 128B+16B -> pad_interval=4 (16*8B), pad_amount=3
    tdm_load_tile_f16((unsigned)(size_t)(void*)Qs,
                      Q + ((size_t)(b * SS + s0)) * HH + head * HDD,
                      (unsigned)HH, (unsigned)(BB * SS), 64u, 16u, (unsigned)HH, 4u, 3u, 1u);
    __builtin_amdgcn_s_wait_tensorcnt(0);
  }
#else
  if (tid < 128) {
    int r = tid >> 3, d = (tid & 7) * 8;
    *(v8h*)&Qs[r * 72 + d] =
        *(const v8h*)&Q[((size_t)(b * SS + s0 + r)) * HH + head * HDD + d];
  }
#endif
  for (int i = tid; i < 16 * HDD; i += 512) Ol[i] = 0.0f;
  __syncthreads();

  // ---- phase 1: scores = Q(16x64) @ K^T(64xS), f32 into LDS ----
  {
    v16h a0 = load_a_frag(Qs, 72);        // K-dim 0..31
    v16h a1 = load_a_frag(Qs + 32, 72);   // K-dim 32..63
    for (int nt = wave; nt < SS / 16; nt += 16) {
      const int sk = nt * 16;
      const _Float16* kb =
          Kt + ((size_t)(b * SS + sk + nloc)) * HH + head * HDD + h16;
      v16h b0 = *(const v16h*)kb;          // k = h16 .. h16+15
      v16h b1 = *(const v16h*)(kb + 32);   // k = 32+h16 ..
      v8f acc = (v8f)0.0f;
      acc = wmma16x16x32(a0, b0, acc);
      acc = wmma16x16x32(a1, b1, acc);
#pragma unroll
      for (int r = 0; r < 8; ++r)
        sc[(r + 8 * half) * SSP + sk + nloc] = acc[r];
    }
  }
  __syncthreads();

  // ---- phase 2+3: per-row exact radix-select of 64th-largest + thresholded softmax ----
  {
    float* rowp = sc + wave * SSP;      // wave w owns query row w

    float m = -3.402823466e+38f;
    for (int j = lane; j < SS; j += 32) m = fmaxf(m, rowp[j]);
    m = wred_max(m);

    unsigned prefix = 0u;
    int need = TOPK;
#pragma unroll 1
    for (int bit = 31; bit >= 0; --bit) {
      const unsigned cand = prefix | (1u << bit);
      const unsigned mask = ~((1u << bit) - 1u);
      int cnt = 0;
      for (int j = lane; j < SS; j += 32) {
        unsigned u = __float_as_uint(rowp[j]);
        unsigned key = (u & 0x80000000u) ? ~u : (u | 0x80000000u);
        cnt += ((key & mask) == cand) ? 1 : 0;
      }
      cnt = wred_isum(cnt);
      if (cnt >= need) prefix = cand; else need -= cnt;
    }
    unsigned tu = (prefix & 0x80000000u) ? (prefix ^ 0x80000000u) : ~prefix;
    const float thr = __uint_as_float(tu);   // exact 64th-largest score

    float s = 0.0f;
    for (int j = lane; j < SS; j += 32) {
      float x = rowp[j];
      float p = (x >= thr) ? __expf(x - m) : 0.0f;
      rowp[j] = p;
      s += p;
    }
    s = wred_sum(s);
    if (lane == 0) rsum[wave] = s;
  }
  __syncthreads();

  // ---- phase 4: O = P(16xS) @ V(Sx64); cross-wave reduce via ds_add_f32 ----
  {
    const int nt = wave & 3;
    v8f acc = (v8f)0.0f;
    for (int c = (wave >> 2); c < SS / 32; c += 4) {   // 16 k-chunks per wave
      const float* pr = sc + nloc * SSP + c * 32 + h16;
      f32x4 x0 = *(const f32x4*)pr;
      f32x4 x1 = *(const f32x4*)(pr + 4);
      f32x4 x2 = *(const f32x4*)(pr + 8);
      f32x4 x3 = *(const f32x4*)(pr + 12);
      v16h a;
#pragma unroll
      for (int q = 0; q < 4; ++q) {
        a[q]      = (_Float16)x0[q];
        a[4 + q]  = (_Float16)x1[q];
        a[8 + q]  = (_Float16)x2[q];
        a[12 + q] = (_Float16)x3[q];
      }
      const _Float16* vb =
          Vt + ((size_t)b * HH + head * HDD + nt * 16 + nloc) * SS + c * 32 + h16;
      v16h bf = *(const v16h*)vb;
      acc = wmma16x16x32(a, bf, acc);
    }
#pragma unroll
    for (int r = 0; r < 8; ++r)
      atomicAdd(&Ol[(r + 8 * half) * HDD + nt * 16 + nloc], acc[r]);
  }
  __syncthreads();

  // ---- phase 5: normalize, emit f16 O tile ----
  for (int i = tid; i < 16 * HDD; i += 512) {
    int r = i >> 6, d = i & 63;
    O16[((size_t)(b * SS + s0 + r)) * HH + head * HDD + d] = (_Float16)(Ol[i] / rsum[r]);
  }
}

// ---------------- host launcher ----------------
extern "C" void kernel_launch(void* const* d_in, const int* in_sizes, int n_in,
                              void* d_out, int out_size, void* d_ws, size_t ws_size,
                              hipStream_t stream) {
  (void)in_sizes; (void)n_in; (void)out_size; (void)ws_size;

  const float* hs = (const float*)d_in[0];
  const float* Wq = (const float*)d_in[1];
  const float* bq = (const float*)d_in[2];
  const float* Wk = (const float*)d_in[3];
  const float* bk = (const float*)d_in[4];
  const float* Wv = (const float*)d_in[5];
  const float* bv = (const float*)d_in[6];
  const float* Wo = (const float*)d_in[7];
  const float* bo = (const float*)d_in[8];
  float* out = (float*)d_out;

  const int M = BB * SS;     // 4096
  const int N = HH;          // 1024
  const int K = HH;          // 1024

  char* ws = (char*)d_ws;
  size_t off = 0;
  auto carve = [&](size_t bytes) -> void* {
    void* p = ws + off;
    off = (off + bytes + 255) & ~(size_t)255;
    return p;
  };
  _Float16* h16  = (_Float16*)carve((size_t)M * HH * 2);
  _Float16* Wq16 = (_Float16*)carve((size_t)HH * HH * 2);
  _Float16* Wk16 = (_Float16*)carve((size_t)HH * HH * 2);
  _Float16* Wv16 = (_Float16*)carve((size_t)HH * HH * 2);
  _Float16* Wo16 = (_Float16*)carve((size_t)HH * HH * 2);
  _Float16* Q16  = (_Float16*)carve((size_t)M * HH * 2);
  _Float16* K16  = (_Float16*)carve((size_t)M * HH * 2);
  _Float16* V16  = (_Float16*)carve((size_t)M * HH * 2);
  _Float16* O16  = (_Float16*)carve((size_t)M * HH * 2);
  _Float16* Vt   = (_Float16*)carve((size_t)M * HH * 2);

  (void)hipFuncSetAttribute((const void*)attn_topk_kernel,
                            hipFuncAttributeMaxDynamicSharedMemorySize,
                            ATTN_SMEM_BYTES);

  // f32 -> f16 conversions (vector loads/stores)
  cvt_f16_kernel<<<1024, 256, 0, stream>>>(hs, h16, M * HH / 4);
  cvt_f16_kernel<<<1024, 256, 0, stream>>>(Wq, Wq16, HH * HH / 4);
  cvt_f16_kernel<<<1024, 256, 0, stream>>>(Wk, Wk16, HH * HH / 4);
  cvt_f16_kernel<<<1024, 256, 0, stream>>>(Wv, Wv16, HH * HH / 4);
  cvt_f16_kernel<<<1024, 256, 0, stream>>>(Wo, Wo16, HH * HH / 4);

  // Q/K/V projections (q-scale folded into Q GEMM: (x@W + b) * HD^-0.5)
  dim3 ggrid(M / 64, N / 128);
  gemm_f16_kernel<<<ggrid, 256, 0, stream>>>(h16, Wq16, bq, 0.125f, Q16, nullptr, M, N, K);
  gemm_f16_kernel<<<ggrid, 256, 0, stream>>>(h16, Wk16, bk, 1.0f,   K16, nullptr, M, N, K);
  gemm_f16_kernel<<<ggrid, 256, 0, stream>>>(h16, Wv16, bv, 1.0f,   V16, nullptr, M, N, K);

  // V transpose for contiguous PV B-fragments
  transpose_v_kernel<<<dim3(SS / 64, HH / 64, BB), 256, 0, stream>>>(V16, Vt);

  // attention with exact top-64 threshold
  attn_topk_kernel<<<BB * NHH * (SS / 16), 512, ATTN_SMEM_BYTES, stream>>>(Q16, K16, Vt, O16);

  // output projection -> f32
  gemm_f16_kernel<<<ggrid, 256, 0, stream>>>(O16, Wo16, bo, 1.0f, nullptr, out, M, N, K);
}
